// ContactNet_81226421502033
// MI455X (gfx1250) — matrix operand order repaired and checked by reading, not compiled, not run on hardware
//
#include <hip/hip_runtime.h>
#include <stdint.h>
#include <stddef.h>

// ---------------------------------------------------------------------------
// ContactNet head for MI455X (gfx1250): per-point 4x (131->128 ReLU MLP ->
// {1,3,3,1}) + grasp-frame postprocessing.
// Layer1 GEMM on v_wmma_f32_16x16x32_bf16 (wave32). W1 is staged once per
// persistent block into LDS, pre-swizzled into per-WMMA-fragment order so
// every A/B fragment is exactly two aligned ds_load_b128 into contiguous
// VGPRs (no v_mov repacking in front of the WMMAs).
// ---------------------------------------------------------------------------

typedef __attribute__((ext_vector_type(16))) __bf16    v16bf;
typedef __attribute__((ext_vector_type(8)))  float     v8f;
typedef __attribute__((ext_vector_type(4)))  float     v4f;
typedef __attribute__((ext_vector_type(4)))  uint32_t  v4u;

constexpr int BATCH = 16, PPTS = 20480;
constexpr int N     = BATCH * PPTS;          // 327680 points
constexpr int CIN   = 131;                   // 3 xyz + 128 features
constexpr int HID   = 128;
constexpr int KP    = 160;                   // K padded to 5 chunks of 32
constexpr int KPAIR = KP / 2;                // bf16 pairs (uint32) per row
constexpr int TROWS = 32;                    // rows per block tile
constexpr int NTILES = N / TROWS;
constexpr int W1_WORDS = 4 * 5 * 8 * 32 * 8; // [head][kc][t][lane][j] = 40960

// LDS layout (bytes)
constexpr size_t SM_W1 = (size_t)W1_WORDS * 4;         // 163840 (swizzled bf16x2)
constexpr size_t SM_B1 = 4 * HID * 4;                  // 2048
constexpr size_t SM_W2 = 8 * HID * 4;                  // 4096 (8 output rows total)
constexpr size_t SM_B2 = 8 * 4;                        // 32
constexpr size_t SM_PF = (size_t)TROWS * KP * 2;       // 10240
constexpr size_t SM_H  = (size_t)8 * 16 * HID * 4;     // 65536
constexpr size_t SM_HO = TROWS * 8 * 4;                // 1024
constexpr size_t SMEM_BYTES = SM_W1 + SM_B1 + SM_W2 + SM_B2 + SM_PF + SM_H + SM_HO;

struct Params {
  const float* feat;          // [N,128]
  const float* pts;           // [N,3]
  const float* W1[4];         // [128,131] each (s, z1, z2, w)
  const float* b1[4];         // [128]
  const float* W2[4];         // [od,128], od = 1,3,3,1
  const float* b2[4];         // [od]
  float*       out;           // pts(3N) | grasps(16N) | s(N) | w(N) | feats(131N)
};

__device__ __forceinline__ uint32_t pack_bf16(float a, float b) {
  union { __bf16 h[2]; uint32_t u; } pk;
  pk.h[0] = (__bf16)a; pk.h[1] = (__bf16)b;
  return pk.u;
}

__device__ __forceinline__ int head_of_oid(int oid) {
  return oid == 0 ? 0 : (oid < 4 ? 1 : (oid < 7 ? 2 : 3));
}
__device__ __forceinline__ int o_of_oid(int oid) {
  return oid == 0 ? 0 : (oid < 4 ? oid - 1 : (oid < 7 ? oid - 4 : 0));
}

__global__ __launch_bounds__(256) void contactnet_kernel(Params prm) {
  extern __shared__ char smem[];
  uint32_t* w1s   = (uint32_t*)smem;                       // swizzled W1 (bf16x2)
  float*    b1l   = (float*)(smem + SM_W1);                // [4][128]
  float*    w2l   = b1l + 4 * HID;                         // [8][128]
  float*    b2l   = w2l + 8 * HID;                         // [8]
  __bf16*   pfl   = (__bf16*)(b2l + 8);                    // [32][KP]
  char*     pfb   = (char*)pfl;
  float*    hbuf  = (float*)(pfb + SM_PF);                 // [8 waves][16][128]
  float*    hout  = hbuf + 8 * 16 * HID;                   // [32][8]

  const int tid     = threadIdx.x;
  const int lane    = tid & 31;                 // wave32
  const int wid     = tid >> 5;                 // 8 waves
  const int head    = wid & 3;                  // head per wave
  const int rowtile = wid >> 2;                 // 0/1: which 16 rows
  const int h16     = lane >> 4;                // lane half
  const int nn      = lane & 15;

  // ---- one-time: stage W1 into LDS pre-swizzled into B-fragment order ------
  // w1s[((head*5+kc)*8+t)*256 + lane*8 + j] = bf16pair of
  //   W1[head][n=t*16+(lane&15)][K = 2*(kc*16 + (lane>>4)*8 + j) + {0,1}]
  for (int idx = tid; idx < W1_WORDS; idx += 256) {
    int j    = idx & 7;
    int ln   = (idx >> 3) & 31;
    int t    = (idx >> 8) & 7;
    int rem  = idx >> 11;              // head*5 + kc
    int kc   = rem % 5;
    int hd   = rem / 5;
    int pidx = kc * 16 + (ln >> 4) * 8 + j;
    int n    = t * 16 + (ln & 15);
    int k0   = 2 * pidx, k1 = k0 + 1;
    const float* W = prm.W1[hd];
    float f0 = (k0 < CIN) ? W[n * CIN + k0] : 0.f;
    float f1 = (k1 < CIN) ? W[n * CIN + k1] : 0.f;
    w1s[idx] = pack_bf16(f0, f1);
  }
  for (int idx = tid; idx < 4 * HID; idx += 256)
    b1l[idx] = prm.b1[idx >> 7][idx & 127];
  for (int idx = tid; idx < 8 * HID; idx += 256) {
    int oid = idx >> 7, j = idx & 127;
    w2l[idx] = prm.W2[head_of_oid(oid)][o_of_oid(oid) * HID + j];
  }
  if (tid < 8) b2l[tid] = prm.b2[head_of_oid(tid)][o_of_oid(tid)];
  __syncthreads();

  float* outPts = prm.out;
  float* outGr  = prm.out + (size_t)3  * N;
  float* outS   = prm.out + (size_t)19 * N;
  float* outW   = prm.out + (size_t)20 * N;
  float* outF   = prm.out + (size_t)21 * N;

  // per-wave constant fragment addresses
  const int rbase  = rowtile * 16 + nn;                 // A-matrix row (m)
  const int rbyte  = rbase * (KPAIR * 4);               // pf row byte offset
  const char* w1b  = (const char*)w1s;

  for (int tile = blockIdx.x; tile < NTILES; tile += gridDim.x) {
    const int row0 = tile * TROWS;

    // ---- stage 1: pf tile -> LDS (bf16) + feats/pts outputs (single read) --
    for (int i = tid; i < TROWS * CIN; i += 256) {
      int r = i / CIN;
      int k = i - r * CIN;
      int gr = row0 + r;
      float v = (k < 3) ? prm.pts[gr * 3 + k] : prm.feat[gr * 128 + (k - 3)];
      pfl[r * KP + k] = (__bf16)v;
      outF[(size_t)gr * CIN + k] = v;
      if (k < 3) outPts[gr * 3 + k] = v;
    }
    for (int i = tid; i < TROWS * (KP - CIN); i += 256) {
      int r = i / (KP - CIN);
      int k = CIN + (i - r * (KP - CIN));
      pfl[r * KP + k] = (__bf16)0.f;
    }
    __syncthreads();

    // ---- stage 2: layer1 GEMM, one (head, 16-row half) per wave ------------
    v8f acc[8] = {};                               // 16x128 f32 accumulators
#pragma unroll
    for (int kc = 0; kc < 5; ++kc) {
      // A fragment: two aligned 16B runs in the row-major pf tile
      union { v16bf v; v4u q[2]; } a;
      const char* ab = pfb + rbyte + kc * 64 + h16 * 32;
      a.q[0] = *(const v4u*)(ab);
      a.q[1] = *(const v4u*)(ab + 16);
#pragma unroll
      for (int t = 0; t < 8; ++t) {
        // B fragment: contiguous 32B per lane in the swizzled W1
        union { v16bf v; v4u q[2]; } b;
        const char* bb = w1b + (((head * 5 + kc) * 8 + t) * 1024) + lane * 32;
        b.q[0] = *(const v4u*)(bb);
        b.q[1] = *(const v4u*)(bb + 16);
        acc[t] = __builtin_amdgcn_wmma_f32_16x16x32_bf16(
            false, a.v, false, b.v, (short)0, acc[t], false, false);
      }
    }

    // ---- stage 3: bias + ReLU, hidden -> LDS -------------------------------
    const int hbase = wid * (16 * HID);
#pragma unroll
    for (int t = 0; t < 8; ++t) {
      int col = t * 16 + nn;
      float bias = b1l[head * HID + col];
#pragma unroll
      for (int r = 0; r < 8; ++r) {
        int mrow = r + 8 * h16;                    // C/D layout: lanes 16-31 -> M+8
        hbuf[hbase + mrow * HID + col] = fmaxf(acc[t][r] + bias, 0.f);
      }
    }
    __syncthreads();

    // ---- stage 4: layer2 — one length-128 dot per thread (32 rows x 8 outs) -
    {
      int row = tid >> 3;
      int oid = tid & 7;
      int hd  = head_of_oid(oid);
      int swid = (row >> 4) * 4 + hd;              // wave that produced this h
      const v4f* hp4 = (const v4f*)(hbuf + swid * (16 * HID) + (row & 15) * HID);
      const v4f* wp4 = (const v4f*)(w2l + oid * HID);
      float s = b2l[oid];
#pragma unroll 8
      for (int j = 0; j < HID / 4; ++j) {
        v4f hv = hp4[j], wv = wp4[j];
        s += hv[0] * wv[0] + hv[1] * wv[1] + hv[2] * wv[2] + hv[3] * wv[3];
      }
      hout[row * 8 + oid] = s;
    }
    __syncthreads();

    // ---- stage 5: grasp-frame postprocessing, one row per lane -------------
    if (tid < TROWS) {
      int gr = row0 + tid;
      const float* ho = hout + tid * 8;
      float s   = ho[0];
      float z1x = ho[1], z1y = ho[2], z1z = ho[3];
      float z2x = ho[4], z2y = ho[5], z2z = ho[6];
      float w   = ho[7];

      float inv1 = rsqrtf(z1x * z1x + z1y * z1y + z1z * z1z);
      z1x *= inv1; z1y *= inv1; z1z *= inv1;                  // base (unit)
      float d = z1x * z2x + z1y * z2y + z1z * z2z;
      z2x -= d * z1x; z2y -= d * z1y; z2z -= d * z1z;
      float inv2 = rsqrtf(z2x * z2x + z2y * z2y + z2z * z2z);
      z2x *= inv2; z2y *= inv2; z2z *= inv2;                  // z2 normalized
      w = fminf(fmaxf(w, -0.08f), 0.08f);

      float inner = z1x * z2x + z1y * z2y + z1z * z2z;        // ~0
      float ax = z2x - inner * z1x, ay = z2y - inner * z1y, az = z2z - inner * z1z;
      float inva = rsqrtf(ax * ax + ay * ay + az * az);
      ax *= inva; ay *= inva; az *= inva;                     // approach (unit)

      float gx = ay * z1z - az * z1y;                         // cross(app, base)
      float gy = az * z1x - ax * z1z;
      float gz = ax * z1y - ay * z1x;
      float invg = rsqrtf(gx * gx + gy * gy + gz * gz);
      gx *= invg; gy *= invg; gz *= invg;

      float px = prm.pts[gr * 3 + 0], py = prm.pts[gr * 3 + 1], pz = prm.pts[gr * 3 + 2];
      float tx = px - 0.1034f * ax + 0.5f * w * z1x;
      float ty = py - 0.1034f * ay + 0.5f * w * z1y;
      float tz = pz - 0.1034f * az + 0.5f * w * z1z;

      float* g = outGr + (size_t)gr * 16;
      g[0]  = z1x; g[1]  = gx; g[2]  = ax; g[3]  = tx;
      g[4]  = z1y; g[5]  = gy; g[6]  = ay; g[7]  = ty;
      g[8]  = z1z; g[9]  = gz; g[10] = az; g[11] = tz;
      g[12] = 0.f; g[13] = 0.f; g[14] = 0.f; g[15] = 1.f;
      outS[gr] = s;
      outW[gr] = fmaxf(w, 0.f);
    }
    // next iteration's first barrier orders pf/hbuf reuse
  }
}

extern "C" void kernel_launch(void* const* d_in, const int* in_sizes, int n_in,
                              void* d_out, int out_size, void* d_ws, size_t ws_size,
                              hipStream_t stream) {
  (void)in_sizes; (void)n_in; (void)d_ws; (void)ws_size; (void)out_size;
  Params prm;
  prm.feat = (const float*)d_in[0];
  prm.pts  = (const float*)d_in[1];
  // d_in[2] = batch ids (only define B, static here)
  for (int h = 0; h < 4; ++h) {
    prm.W1[h] = (const float*)d_in[3 + 4 * h];
    prm.b1[h] = (const float*)d_in[4 + 4 * h];
    prm.W2[h] = (const float*)d_in[5 + 4 * h];
    prm.b2[h] = (const float*)d_in[6 + 4 * h];
  }
  prm.out = (float*)d_out;

  (void)hipFuncSetAttribute(reinterpret_cast<const void*>(contactnet_kernel),
                            hipFuncAttributeMaxDynamicSharedMemorySize,
                            (int)SMEM_BYTES);

  dim3 grid(640), block(256);
  contactnet_kernel<<<grid, block, SMEM_BYTES, stream>>>(prm);
}
